// MultiFlowCondAdaptRouter_62878321214034
// MI455X (gfx1250) — matrix operand order repaired and checked by reading, not compiled, not run on hardware
//
#include <hip/hip_runtime.h>
#include <stdint.h>

// ---------------- problem constants ----------------
#define M_EXP   8
#define B_SZ    8
#define N_TOK   2048
#define D_MODEL 512
#define D_FF    2048

// ---------------- tiling constants -----------------
#define TN   64            // token rows per workgroup
#define FC   512           // DFF chunk width kept in LDS
#define NKS  16            // 512 / 32 k-steps
#define XSTR 520           // XB / H row stride (bf16 elems): 512 + 8 pad (bank-conflict free)
#define WSTR 40            // WT row stride (bf16 elems): 32 + 8 pad (16B aligned, conflict free)
#define WTBUF_BYTES (FC * WSTR * 2)            // 40960 per panel buffer

// LDS layout (bytes)
#define XB_OFF 0
#define H_OFF  (TN * XSTR * 2)                 // 66560
#define WT_OFF (H_OFF + TN * XSTR * 2)         // 133120  (двойной buffer: 2 x 40960)
#define SC_OFF (WT_OFF + 2 * WTBUF_BYTES)      // 215040
#define SH_OFF (SC_OFF + 2048)                 // 217088
#define LDS_BYTES (SH_OFF + 2048)              // 219136 (< 320 KB/WGP)

#define NSTEPS 128                             // 4 fc * 2 phases * 16 k-steps

typedef __attribute__((ext_vector_type(16))) __bf16 v16bf;
typedef __attribute__((ext_vector_type(8)))  float  v8f;
typedef __attribute__((ext_vector_type(4)))  unsigned int u32x4;
typedef __attribute__((ext_vector_type(8)))  int i32x8;
typedef __attribute__((ext_vector_type(4)))  int i32x4;

#define HAS_TDM __has_builtin(__builtin_amdgcn_tensor_load_to_lds)

union Frag { uint4 q[2]; v16bf v; };

__device__ __forceinline__ unsigned short f2bf(float f) {
  unsigned int u = __float_as_uint(f);
  u += 0x7fffu + ((u >> 16) & 1u);   // round-to-nearest-even
  return (unsigned short)(u >> 16);
}

__device__ __forceinline__ float gelu_tanh(float x) {
  float x3 = x * x * x;
  return 0.5f * x * (1.0f + tanhf(0.7978845608028654f * (x + 0.044715f * x3)));
}

// A-operand (16x32 bf16, M rows across lanes, K chunks per ISA table):
// lane<16: K = [kb..kb+7],[kb+16..kb+23]; lane>=16: K = [kb+8..kb+15],[kb+24..kb+31]
__device__ __forceinline__ v16bf load_a(const char* base, int mt, int kb, int lane) {
  int row = mt * 16 + (lane & 15);
  int off = row * (XSTR * 2) + kb * 2 + (lane >> 4) * 16;
  Frag f;
  f.q[0] = *(const uint4*)(base + off);
  f.q[1] = *(const uint4*)(base + off + 32);
  return f.v;
}

// B-operand (32x16 bf16): lane holds column n = lane&15; lanes 0-15: K 0-7,16-23, lanes 16-31: K 8-15,24-31
__device__ __forceinline__ v16bf load_b(const char* wt, int ct, int lane) {
  int col = ct * 16 + (lane & 15);
  int off = col * (WSTR * 2) + (lane >> 4) * 16;
  Frag f;
  f.q[0] = *(const uint4*)(wt + off);
  f.q[1] = *(const uint4*)(wt + off + 32);
  return f.v;
}

#if HAS_TDM
// ----- Tensor Data Mover descriptor construction (D#) -----
// Group 0: count=1 | lds_addr[63:32] | global_addr[120:64] | type=2
__device__ __forceinline__ u32x4 tdm_g0(unsigned lds_addr, unsigned long long gaddr) {
  u32x4 g;
  g[0] = 1u;                                          // count=1, user descriptor
  g[1] = lds_addr;
  g[2] = (unsigned)(gaddr & 0xffffffffu);
  g[3] = (unsigned)((gaddr >> 32) & 0x01ffffffu) | (2u << 30);   // type=2 ("image")
  return g;
}

// Group 1: data_size=2B, pad 4 dwords (16B) every 16 dwords (64B) -> 80B LDS row stride
__device__ __forceinline__ i32x8 tdm_g1(unsigned td0, unsigned td1,
                                        unsigned tile0, unsigned tile1,
                                        unsigned long long stride0) {
  i32x8 g;
  g[0] = (1 << 16)        // data_size = 1 -> 2 bytes
       | (1 << 20)        // pad_enable
       | (3 << 22)        // pad_interval: 8<<3 = 64 bytes
       | (3 << 25);       // pad_amount: 4 dwords = 16 bytes
  g[1] = (int)((td0 & 0xffffu) << 16);                                  // tensor_dim0[15:0]
  g[2] = (int)(((td0 >> 16) & 0xffffu) | ((td1 & 0xffffu) << 16));      // td0 hi | td1 lo
  g[3] = (int)(((td1 >> 16) & 0xffffu) | ((tile0 & 0xffffu) << 16));    // td1 hi | tile_dim0
  g[4] = (int)(tile1 & 0xffffu);                                        // tile_dim1 (tile_dim2=0)
  g[5] = (int)(unsigned)(stride0 & 0xffffffffu);                        // tensor_dim0_stride lo
  g[6] = (int)((stride0 >> 32) & 0xffffu);                              // stride hi (dim1_stride=0)
  g[7] = 0;
  return g;
}

__device__ __forceinline__ void tdm_issue(u32x4 g0, i32x8 g1) {
  i32x4 z4 = {0, 0, 0, 0};
#if defined(__clang_major__) && __clang_major__ >= 23
  i32x8 z8 = {0, 0, 0, 0, 0, 0, 0, 0};
  __builtin_amdgcn_tensor_load_to_lds(g0, g1, z4, z4, z8, 0);
#else
  __builtin_amdgcn_tensor_load_to_lds(g0, g1, z4, z4, 0);
#endif
}

// One weight panel = 2D tile [tile1=512 rows x tile0=32 elems], row pitch = matrix row length.
__device__ __forceinline__ void issue_panel(int step, int m,
                                            const unsigned short* __restrict__ w1t,
                                            const unsigned short* __restrict__ w2t,
                                            unsigned wt_lds_base) {
  const int fc = step >> 5;
  const int ph = (step >> 4) & 1;
  const int ks = step & 15;
  const unsigned lds = wt_lds_base + (unsigned)(step & 1) * WTBUF_BYTES;
  unsigned long long ga;
  i32x8 g1;
  if (ph == 0) {   // W1T panel: rows f = fc*512.., cols k = ks*32.., pitch D_MODEL
    const unsigned short* p = w1t + (size_t)m * D_FF * D_MODEL
                                  + (size_t)(fc * FC) * D_MODEL + (size_t)ks * 32;
    ga = (unsigned long long)(size_t)p;
    g1 = tdm_g1(D_MODEL, D_FF, 32, FC, D_MODEL);
  } else {         // W2T panel: rows d = 0..511, cols f = fc*512+ks*32.., pitch D_FF
    const unsigned short* p = w2t + (size_t)m * D_MODEL * D_FF
                                  + (size_t)(fc * FC) + (size_t)ks * 32;
    ga = (unsigned long long)(size_t)p;
    g1 = tdm_g1(D_FF, D_MODEL, 32, FC, D_FF);
  }
  tdm_issue(tdm_g0(lds, ga), g1);
}
#else
__device__ __forceinline__ void stage_panel_manual(int step, int m, int tid, char* WT,
                                                   const unsigned short* __restrict__ w1t,
                                                   const unsigned short* __restrict__ w2t) {
  const int fc = step >> 5;
  const int ph = (step >> 4) & 1;
  const int ks = step & 15;
  const unsigned short* src = (ph == 0)
      ? w1t + (size_t)m * D_FF * D_MODEL + (size_t)(fc * FC + tid) * D_MODEL + ks * 32
      : w2t + (size_t)m * D_MODEL * D_FF + (size_t)tid * D_FF + fc * FC + ks * 32;
  char* dst = WT + (step & 1) * WTBUF_BYTES + tid * (WSTR * 2);
  uint4 q0 = *(const uint4*)(src);
  uint4 q1 = *(const uint4*)(src + 8);
  uint4 q2 = *(const uint4*)(src + 16);
  uint4 q3 = *(const uint4*)(src + 24);
  *(uint4*)(dst)      = q0;
  *(uint4*)(dst + 16) = q1;
  *(uint4*)(dst + 32) = q2;
  *(uint4*)(dst + 48) = q3;
}
#endif

// ---------- prep: [Mb][R][C] f32 -> [Mb][C][R] bf16 (tiled transpose + convert) ----------
__global__ void convert_transpose_kernel(const float* __restrict__ in,
                                         unsigned short* __restrict__ out,
                                         int R, int C) {
  __shared__ float tile[32][33];
  const int mb = blockIdx.z;
  const float* src = in + (size_t)mb * R * C;
  unsigned short* dst = out + (size_t)mb * R * C;
  const int c0 = blockIdx.x * 32, r0 = blockIdx.y * 32;
  const int tx = threadIdx.x, ty = threadIdx.y;   // 32 x 8
  #pragma unroll
  for (int i = 0; i < 4; ++i)
    tile[ty + i * 8][tx] = src[(size_t)(r0 + ty + i * 8) * C + c0 + tx];
  __syncthreads();
  #pragma unroll
  for (int i = 0; i < 4; ++i)
    dst[(size_t)(c0 + ty + i * 8) * R + r0 + tx] = f2bf(tile[tx][ty + i * 8]);
}

// ---------- routed cond modulation ----------
__global__ void mod_kernel(const float* __restrict__ cond,
                           const long long* __restrict__ route,
                           const float* __restrict__ Wc,
                           const float* __restrict__ bc,
                           float* __restrict__ wsMod) {
  const int b = blockIdx.y;
  const int j = blockIdx.x * blockDim.x + threadIdx.x;   // 0..1023
  const int m = (int)route[b];
  float s = bc[m * (2 * D_MODEL) + j];
  const float* w  = Wc + (size_t)m * D_MODEL * (2 * D_MODEL) + j;
  const float* cv = cond + b * D_MODEL;
  for (int c = 0; c < D_MODEL; ++c) s += cv[c] * w[(size_t)c * (2 * D_MODEL)];
  wsMod[b * (2 * D_MODEL) + j] = s;
}

// ---------- fused FiLM -> GEMM1 -> GeLU -> GEMM2 (bf16 WMMA + TDM staging) ----------
__global__ __launch_bounds__(512)
void moe_film_mlp_kernel(const float* __restrict__ x,
                         const long long* __restrict__ route,
                         const float* __restrict__ b1g,
                         const float* __restrict__ b2g,
                         const float* __restrict__ wsMod,
                         const unsigned short* __restrict__ w1t,   // [M][DFF][D] bf16
                         const unsigned short* __restrict__ w2t,   // [M][D][DFF] bf16
                         float* __restrict__ out) {
  extern __shared__ char smem[];
  char*  XB = smem + XB_OFF;
  char*  H  = smem + H_OFF;
  char*  WT = smem + WT_OFF;
  float* SC = (float*)(smem + SC_OFF);
  float* SH = (float*)(smem + SH_OFF);

  const int tid  = threadIdx.x;
  const int lane = tid & 31;
  const int w    = tid >> 5;        // wave 0..15
  const int mgrp = w & 1;           // owns M-tiles {2*mgrp, 2*mgrp+1}
  const int cgrp = w >> 1;          // owns column tiles cgrp*4 .. cgrp*4+3
  const int b    = blockIdx.y;
  const int n0   = blockIdx.x * TN;
  const int m    = (int)route[b];

  // scale/shift into LDS (scale1 = 1 + scale)
  SC[tid] = 1.0f + wsMod[b * (2 * D_MODEL) + tid];
  SH[tid] = wsMod[b * (2 * D_MODEL) + D_MODEL + tid];

#if HAS_TDM
  const unsigned wt_lds_base = (unsigned)(size_t)(smem + WT_OFF);
  if (w == 0) issue_panel(0, m, w1t, w2t, wt_lds_base);   // prefetch first panel via TDM
#endif
  __syncthreads();

  // stage FiLM-modulated x rows as bf16 into XB
  {
    const int r = tid >> 3, s = tid & 7;
    const float* xr = x + (size_t)(b * N_TOK + n0 + r) * D_MODEL + s * 64;
    char* dst = XB + r * (XSTR * 2) + s * 128;
    #pragma unroll
    for (int j = 0; j < 8; ++j) {
      float4 v0 = *(const float4*)(xr + j * 8);
      float4 v1 = *(const float4*)(xr + j * 8 + 4);
      const int c = s * 64 + j * 8;
      union { unsigned short h[8]; uint4 q; } pk;
      pk.h[0] = f2bf(fmaf(v0.x, SC[c + 0], SH[c + 0]));
      pk.h[1] = f2bf(fmaf(v0.y, SC[c + 1], SH[c + 1]));
      pk.h[2] = f2bf(fmaf(v0.z, SC[c + 2], SH[c + 2]));
      pk.h[3] = f2bf(fmaf(v0.w, SC[c + 3], SH[c + 3]));
      pk.h[4] = f2bf(fmaf(v1.x, SC[c + 4], SH[c + 4]));
      pk.h[5] = f2bf(fmaf(v1.y, SC[c + 5], SH[c + 5]));
      pk.h[6] = f2bf(fmaf(v1.z, SC[c + 6], SH[c + 6]));
      pk.h[7] = f2bf(fmaf(v1.w, SC[c + 7], SH[c + 7]));
      *(uint4*)(dst + j * 16) = pk.q;
    }
  }

  v8f acc1[2][4], acc2[2][4];
  #pragma unroll
  for (int i = 0; i < 2; ++i)
    #pragma unroll
    for (int t = 0; t < 4; ++t)
      #pragma unroll
      for (int e = 0; e < 8; ++e) acc2[i][t][e] = 0.0f;

  // 128 flat steps: step = fc*32 + phase*16 + ks ; phase 0 = GEMM1, phase 1 = GEMM2
  for (int step = 0; step < NSTEPS; ++step) {
    const int fc = step >> 5;
    const int ph = (step >> 4) & 1;
    const int ks = step & 15;

#if HAS_TDM
    if (w == 0) {
      if (step + 1 < NSTEPS) {
        issue_panel(step + 1, m, w1t, w2t, wt_lds_base);   // prefetch next panel
        __builtin_amdgcn_s_wait_tensorcnt((short)1);       // current panel landed
      } else {
        __builtin_amdgcn_s_wait_tensorcnt((short)0);
      }
    }
    __syncthreads();
#else
    __syncthreads();
    stage_panel_manual(step, m, tid, WT, w1t, w2t);
    __syncthreads();
#endif

    const char* wtb = WT + (step & 1) * WTBUF_BYTES;

    if (ph == 0) {
      if (ks == 0) {
        #pragma unroll
        for (int i = 0; i < 2; ++i)
          #pragma unroll
          for (int t = 0; t < 4; ++t)
            #pragma unroll
            for (int e = 0; e < 8; ++e) acc1[i][t][e] = 0.0f;
      }
      v16bf a0 = load_a(XB, 2 * mgrp,     ks * 32, lane);
      v16bf a1 = load_a(XB, 2 * mgrp + 1, ks * 32, lane);
      #pragma unroll
      for (int t = 0; t < 4; ++t) {
        v16bf bb = load_b(wtb, cgrp * 4 + t, lane);
        acc1[0][t] = __builtin_amdgcn_wmma_f32_16x16x32_bf16(false, a0, false, bb,
                       (short)0, acc1[0][t], false, false);
        acc1[1][t] = __builtin_amdgcn_wmma_f32_16x16x32_bf16(false, a1, false, bb,
                       (short)0, acc1[1][t], false, false);
      }
      if (ks == 15) {
        // bias + GeLU -> H (bf16)
        #pragma unroll
        for (int i = 0; i < 2; ++i) {
          const int mt = 2 * mgrp + i;
          #pragma unroll
          for (int t = 0; t < 4; ++t) {
            const int ct   = cgrp * 4 + t;
            const int coln = ct * 16 + (lane & 15);
            const float bias = b1g[m * D_FF + fc * FC + coln];
            #pragma unroll
            for (int e = 0; e < 8; ++e) {
              const int row = mt * 16 + (lane >> 4) * 8 + e;
              const float hv = gelu_tanh(acc1[i][t][e] + bias);
              *(unsigned short*)(H + row * (XSTR * 2) + coln * 2) = f2bf(hv);
            }
          }
        }
      }
    } else {
      v16bf a0 = load_a(H, 2 * mgrp,     ks * 32, lane);
      v16bf a1 = load_a(H, 2 * mgrp + 1, ks * 32, lane);
      #pragma unroll
      for (int t = 0; t < 4; ++t) {
        v16bf bb = load_b(wtb, cgrp * 4 + t, lane);
        acc2[0][t] = __builtin_amdgcn_wmma_f32_16x16x32_bf16(false, a0, false, bb,
                       (short)0, acc2[0][t], false, false);
        acc2[1][t] = __builtin_amdgcn_wmma_f32_16x16x32_bf16(false, a1, false, bb,
                       (short)0, acc2[1][t], false, false);
      }
    }
    __syncthreads();   // panel buffer free for TDM two steps ahead
  }

  // ---- epilogue: bias2 + store f32 output ----
  #pragma unroll
  for (int i = 0; i < 2; ++i) {
    const int mt = 2 * mgrp + i;
    #pragma unroll
    for (int t = 0; t < 4; ++t) {
      const int dt  = cgrp * 4 + t;
      const int col = dt * 16 + (lane & 15);
      const float bias = b2g[m * D_MODEL + col];
      #pragma unroll
      for (int e = 0; e < 8; ++e) {
        const int row = mt * 16 + (lane >> 4) * 8 + e;
        out[(size_t)(b * N_TOK + n0 + row) * D_MODEL + col] = acc2[i][t][e] + bias;
      }
    }
  }
}

extern "C" void kernel_launch(void* const* d_in, const int* in_sizes, int n_in,
                              void* d_out, int out_size, void* d_ws, size_t ws_size,
                              hipStream_t stream) {
  (void)in_sizes; (void)n_in; (void)out_size; (void)ws_size;
  const float*     x      = (const float*)d_in[0];
  const float*     cond   = (const float*)d_in[1];
  const long long* route  = (const long long*)d_in[2];
  const float*     W_cond = (const float*)d_in[3];
  const float*     b_cond = (const float*)d_in[4];
  const float*     W1     = (const float*)d_in[5];
  const float*     b1     = (const float*)d_in[6];
  const float*     W2     = (const float*)d_in[7];
  const float*     b2     = (const float*)d_in[8];
  float* out = (float*)d_out;

  char* ws = (char*)d_ws;
  float* wsMod = (float*)ws;                                   // 32 KB
  unsigned short* w1t = (unsigned short*)(ws + 32768);         // 16 MB
  unsigned short* w2t = w1t + (size_t)M_EXP * D_FF * D_MODEL;  // 16 MB

  convert_transpose_kernel<<<dim3(D_FF / 32, D_MODEL / 32, M_EXP), dim3(32, 8), 0, stream>>>(
      W1, w1t, D_MODEL, D_FF);
  convert_transpose_kernel<<<dim3(D_MODEL / 32, D_FF / 32, M_EXP), dim3(32, 8), 0, stream>>>(
      W2, w2t, D_FF, D_MODEL);

  mod_kernel<<<dim3((2 * D_MODEL) / 256, B_SZ), 256, 0, stream>>>(cond, route, W_cond, b_cond, wsMod);

  moe_film_mlp_kernel<<<dim3(N_TOK / TN, B_SZ), 512, LDS_BYTES, stream>>>(
      x, route, b1, b2, wsMod, w1t, w2t, out);
}
